// BiLSTM_CRF_84086869721885
// MI455X (gfx1250) — matrix operand order repaired and checked by reading, not compiled, not run on hardware
//
#include <hip/hip_runtime.h>
#include <hip/hip_bf16.h>
#include <math.h>

// ---------------- problem constants ----------------
#define Tn    256
#define Bn    512
#define EMBn  50
#define Hn    100          // per-direction hidden
#define G4    400          // 4*Hn gates
#define HIDn  200
#define Kn    32
#define START_TAG 30
#define END_TAG   31
#define NEGV  (-10000.0f)

// WMMA tiling for the recurrent GEMM: A=[16 x 160] f16 ([x(50)|h(100)|pad]),
// B=[160 x 512] f16 ([W_ih|W_hh|pad] transposed, N padded 400->512), C f32.
#define KPAD  160          // 5 k-iters of 32
#define NPAD  512          // 32 n-tiles of 16

typedef __attribute__((ext_vector_type(16))) _Float16 v16h;
typedef __attribute__((ext_vector_type(8)))  _Float16 v8h;
typedef __attribute__((ext_vector_type(8)))  float    v8f;

__device__ __forceinline__ float sigf(float x) { return 1.0f / (1.0f + expf(-x)); }

// ---------------------------------------------------------------------------
// Kernel 1: fused embedding gather + bidirectional LSTM scan.
// grid = 64 blocks (32 batch-tiles x 2 directions), 256 threads (8 waves).
// Per step, gates[16,400] = [x|h] @ [W_ih|W_hh]^T via v_wmma_f32_16x16x32_f16.
// W lives in LDS; B fragments are re-read from LDS each step (ds_load_b128)
// to keep VGPR pressure low enough to avoid scratch spills on the serial path.
// ---------------------------------------------------------------------------
__global__ void lstm_scan_kernel(const int* __restrict__ sentence,
                                 const int* __restrict__ sen_len,
                                 const float* __restrict__ emb,
                                 const float* __restrict__ Wf_ih,
                                 const float* __restrict__ Wf_hh,
                                 const float* __restrict__ bf,
                                 const float* __restrict__ Wb_ih,
                                 const float* __restrict__ Wb_hh,
                                 const float* __restrict__ bb,
                                 _Float16* __restrict__ h_out)  // [T,B,200] f16
{
    extern __shared__ char smem[];
    _Float16* Wl    = (_Float16*)smem;                                   // 512*160 f16 = 160KB
    float*    gates = (float*)(smem + NPAD * KPAD * 2);                  // 16*512 f32 = 32KB
    _Float16* hA    = (_Float16*)(smem + NPAD * KPAD * 2 + 16 * NPAD * 4); // 16*160 f16
    float*    cS    = (float*)((char*)hA + 16 * KPAD * 2);               // 16*100 f32
    float*    bl    = cS + 16 * Hn;                                      // 400 f32

    const int tid   = threadIdx.x;
    const int dir   = blockIdx.x & 1;       // 0 = forward, 1 = backward
    const int btile = blockIdx.x >> 1;      // 0..31

    const float* Wih  = dir ? Wb_ih : Wf_ih;
    const float* Whh  = dir ? Wb_hh : Wf_hh;
    const float* bias = dir ? bb    : bf;

    // Stage combined weight matrix [n=0..511][k=0..159] into LDS as f16.
    for (int i = tid; i < NPAD * KPAD; i += 256) {
        int n = i / KPAD, k = i % KPAD;
        float v = 0.0f;
        if (n < G4) {
            if (k < EMBn)            v = Wih[n * EMBn + k];
            else if (k < EMBn + Hn)  v = Whh[n * Hn + (k - EMBn)];
        }
        Wl[n * KPAD + k] = (_Float16)v;
    }
    for (int i = tid; i < G4; i += 256)      bl[i] = bias[i];
    for (int i = tid; i < 16 * KPAD; i += 256) hA[i] = (_Float16)0.0f;
    for (int i = tid; i < 16 * Hn; i += 256)   cS[i] = 0.0f;
    __syncthreads();

    const int wave  = tid >> 5;
    const int lane  = tid & 31;
    const int mrow  = lane & 15;
    const int khalf = lane >> 4;

    // Per-lane LDS base pointers for the B fragments of this wave's 4 n-tiles.
    const _Float16* wb0 = Wl + ((wave +  0) * 16 + mrow) * KPAD + khalf * 16;
    const _Float16* wb1 = Wl + ((wave +  8) * 16 + mrow) * KPAD + khalf * 16;
    const _Float16* wb2 = Wl + ((wave + 16) * 16 + mrow) * KPAD + khalf * 16;
    const _Float16* wb3 = Wl + ((wave + 24) * 16 + mrow) * KPAD + khalf * 16;
    const _Float16* wbs[4] = { wb0, wb1, wb2, wb3 };

    for (int t = 0; t < Tn; ++t) {
        const int ts = dir ? (Tn - 1 - t) : t;

        // Phase 1: embedding gather -> f16 x-part of the A tile.
        for (int e = tid; e < 16 * EMBn; e += 256) {
            int m = e / EMBn, k = e % EMBn;
            int b = btile * 16 + m;
            int word = sentence[b * Tn + ts];
            hA[m * KPAD + k] = (_Float16)emb[word * EMBn + k];
        }
        if (t + 1 < Tn) {
            int tnext = dir ? (Tn - 2 - t) : (t + 1);
            __builtin_prefetch((const void*)(sentence + (btile * 16) * Tn + tnext), 0, 1);
        }
        __syncthreads();

        // Phase 2: WMMA. A fragments in registers; B fragments streamed from LDS.
        v16h af[5];
#pragma unroll
        for (int ki = 0; ki < 5; ++ki) {
            const v8h lo = *(const v8h*)(hA + mrow * KPAD + ki * 32 + khalf * 8);
            const v8h hi = *(const v8h*)(hA + mrow * KPAD + ki * 32 + 16 + khalf * 8);
            v16h a;
#pragma unroll
            for (int j = 0; j < 8; ++j) { a[j] = lo[j]; a[8 + j] = hi[j]; }
            af[ki] = a;
        }
#pragma unroll
        for (int i = 0; i < 4; ++i) {
            const _Float16* wb = wbs[i];
            v8f c = {};
#pragma unroll
            for (int ki = 0; ki < 5; ++ki) {
                const v8h blo = *(const v8h*)(wb + ki * 32 - khalf * 8);
                const v8h bhi = *(const v8h*)(wb + ki * 32 - khalf * 8 + 16);
                v16h b;
#pragma unroll
                for (int j = 0; j < 8; ++j) { b[j] = blo[j]; b[8 + j] = bhi[j]; }
                c = __builtin_amdgcn_wmma_f32_16x16x32_f16(
                        false, af[ki], false, b, (short)0, c, false, false);
            }
            int n = (wave + i * 8) * 16 + mrow;
            int mb = khalf * 8;
#pragma unroll
            for (int r = 0; r < 8; ++r)
                gates[(mb + r) * NPAD + n] = c[r];
        }
        __syncthreads();

        // Phase 3: LSTM cell elementwise + mask; feed h back into A tile.
        for (int e = tid; e < 16 * Hn; e += 256) {
            int m = e / Hn, j = e % Hn;
            int b = btile * 16 + m;
            float gi = gates[m * NPAD + j]            + bl[j];
            float gf = gates[m * NPAD + Hn + j]       + bl[Hn + j];
            float gg = gates[m * NPAD + 2 * Hn + j]   + bl[2 * Hn + j];
            float go = gates[m * NPAD + 3 * Hn + j]   + bl[3 * Hn + j];
            float cold = cS[m * Hn + j];
            float c2 = sigf(gf) * cold + sigf(gi) * tanhf(gg);
            float h2 = sigf(go) * tanhf(c2);
            float mt = (ts < sen_len[b]) ? 1.0f : 0.0f;
            h2 *= mt; c2 *= mt;
            cS[m * Hn + j] = c2;
            hA[m * KPAD + EMBn + j] = (_Float16)h2;
            h_out[((size_t)ts * Bn + b) * HIDn + dir * Hn + j] = (_Float16)h2;
        }
        __syncthreads();
    }
}

// ---------------------------------------------------------------------------
// Kernel 2: LayerNorm over HID=200 + linear to K=32 feats. Wave per row.
// grid = T*B/8 blocks of 256 threads.
// ---------------------------------------------------------------------------
__global__ void ln_linear_kernel(const _Float16* __restrict__ h_out,
                                 const float* __restrict__ gamma,
                                 const float* __restrict__ beta,
                                 const float* __restrict__ W_lin,  // [32,200]
                                 float* __restrict__ feats)        // [T,B,32]
{
    __shared__ float Wl[Kn * HIDn];
    __shared__ float gb[2 * HIDn];
    __shared__ float hrow[8 * HIDn];

    const int tid = threadIdx.x;
    for (int i = tid; i < Kn * HIDn; i += 256) Wl[i] = W_lin[i];
    for (int i = tid; i < HIDn; i += 256) { gb[i] = gamma[i]; gb[HIDn + i] = beta[i]; }
    __syncthreads();

    const int wave = tid >> 5, lane = tid & 31;
    const size_t row = (size_t)blockIdx.x * 8 + wave;
    const _Float16* hp = h_out + row * HIDn;
    float* hl = hrow + wave * HIDn;

    float s = 0.0f, s2 = 0.0f;
    for (int j = lane; j < HIDn; j += 32) {
        float v = (float)hp[j];
        hl[j] = v; s += v; s2 += v * v;
    }
#pragma unroll
    for (int off = 16; off > 0; off >>= 1) {
        s  += __shfl_xor(s,  off, 32);
        s2 += __shfl_xor(s2, off, 32);
    }
    float mu   = s / (float)HIDn;
    float var  = s2 / (float)HIDn - mu * mu;
    float rstd = rsqrtf(var + 1e-5f);
    __syncthreads();

    float acc = 0.0f;
    for (int j = 0; j < HIDn; ++j) {
        float hn = (hl[j] - mu) * rstd * gb[j] + gb[HIDn + j];
        acc += hn * Wl[lane * HIDn + j];
    }
    feats[row * Kn + lane] = acc;
}

// ---------------------------------------------------------------------------
// Kernel 3: CRF forward algorithm + gold-path score. Wave per batch element,
// lane = tag. grid = B/8 blocks of 256 threads.
// ---------------------------------------------------------------------------
__global__ void crf_kernel(const int* __restrict__ tags,
                           const int* __restrict__ sen_len,
                           const float* __restrict__ trans,   // [32,32]
                           const float* __restrict__ feats,   // [T,B,32]
                           float* __restrict__ res)           // [B]
{
    __shared__ float tl[Kn * Kn];
    const int tid = threadIdx.x;
    for (int i = tid; i < Kn * Kn; i += 256) tl[i] = trans[i];
    __syncthreads();

    const int wave = tid >> 5, lane = tid & 31;
    const int b = blockIdx.x * 8 + wave;
    const int len = sen_len[b];

    float alpha = (lane == START_TAG) ? 0.0f : NEGV;

    for (int t = 0; t < Tn; ++t) {
        if (t < len) {   // wave-uniform predicate
            float ft = feats[((size_t)t * Bn + b) * Kn + lane];
            float vmax = -3.4e38f;
#pragma unroll 4
            for (int i = 0; i < Kn; ++i) {
                float ai = __shfl(alpha, i, 32);
                float v  = ai + tl[i * Kn + lane];
                vmax = fmaxf(vmax, v);
            }
            float ssum = 0.0f;
#pragma unroll 4
            for (int i = 0; i < Kn; ++i) {
                float ai = __shfl(alpha, i, 32);
                float v  = ai + tl[i * Kn + lane];
                ssum += expf(v - vmax);
            }
            alpha = vmax + logf(ssum) + ft;
        }
    }

    // total = logsumexp_j(alpha_j + trans[j][END])
    float v = alpha + tl[lane * Kn + END_TAG];
    float vmax = v;
#pragma unroll
    for (int off = 16; off > 0; off >>= 1) vmax = fmaxf(vmax, __shfl_xor(vmax, off, 32));
    float e = expf(v - vmax);
#pragma unroll
    for (int off = 16; off > 0; off >>= 1) e += __shfl_xor(e, off, 32);
    float total = vmax + logf(e);

    // gold-path score, timesteps distributed over lanes
    float acc = 0.0f;
    for (int t = lane; t < Tn; t += 32) {
        if (t < len) {
            int prev = (t == 0) ? START_TAG : tags[b * Tn + (t - 1)];
            int cur  = tags[b * Tn + t];
            acc += tl[prev * Kn + cur] + feats[((size_t)t * Bn + b) * Kn + cur];
        }
    }
#pragma unroll
    for (int off = 16; off > 0; off >>= 1) acc += __shfl_xor(acc, off, 32);

    if (lane == 0) {
        int last = tags[b * Tn + (len - 1)];
        float real = acc + tl[last * Kn + END_TAG];
        res[b] = total - real;
    }
}

// ---------------------------------------------------------------------------
// Kernel 4: deterministic mean over B=512.
// ---------------------------------------------------------------------------
__global__ void reduce_kernel(const float* __restrict__ res, float* __restrict__ out)
{
    __shared__ float s[Bn];
    const int tid = threadIdx.x;
    s[tid] = res[tid];
    __syncthreads();
    for (int off = Bn / 2; off > 0; off >>= 1) {
        if (tid < off) s[tid] += s[tid + off];
        __syncthreads();
    }
    if (tid == 0) out[0] = s[0] / (float)Bn;
}

// ---------------------------------------------------------------------------
extern "C" void kernel_launch(void* const* d_in, const int* in_sizes, int n_in,
                              void* d_out, int out_size, void* d_ws, size_t ws_size,
                              hipStream_t stream) {
    const int*   sentence = (const int*)  d_in[0];
    const int*   tags     = (const int*)  d_in[1];
    const int*   sen_len  = (const int*)  d_in[2];
    const float* emb      = (const float*)d_in[3];
    const float* Wf_ih    = (const float*)d_in[4];
    const float* Wf_hh    = (const float*)d_in[5];
    const float* bf       = (const float*)d_in[6];
    const float* Wb_ih    = (const float*)d_in[7];
    const float* Wb_hh    = (const float*)d_in[8];
    const float* bb       = (const float*)d_in[9];
    const float* gamma    = (const float*)d_in[10];
    const float* beta     = (const float*)d_in[11];
    const float* W_lin    = (const float*)d_in[12];
    const float* trans    = (const float*)d_in[13];

    // Workspace layout.
    char* ws = (char*)d_ws;
    _Float16* h_out = (_Float16*)ws;                                   // T*B*200 f16  = 52,428,800 B
    float*    feats = (float*)(ws + (size_t)Tn * Bn * HIDn * 2);       // T*B*32  f32  = 16,777,216 B
    float*    res   = (float*)(ws + (size_t)Tn * Bn * HIDn * 2
                                  + (size_t)Tn * Bn * Kn * 4);         // B f32

    // Dynamic LDS for the scan kernel: W(160KB)+gates(32KB)+A-tile+c+bias.
    const size_t lds_bytes = (size_t)NPAD * KPAD * 2   // Wl
                           + (size_t)16 * NPAD * 4     // gates
                           + (size_t)16 * KPAD * 2     // hA
                           + (size_t)16 * Hn * 4       // cS
                           + (size_t)G4 * 4;           // bias

    lstm_scan_kernel<<<64, 256, lds_bytes, stream>>>(
        sentence, sen_len, emb, Wf_ih, Wf_hh, bf, Wb_ih, Wb_hh, bb, h_out);

    ln_linear_kernel<<<(Tn * Bn) / 8, 256, 0, stream>>>(
        h_out, gamma, beta, W_lin, feats);

    crf_kernel<<<Bn / 8, 256, 0, stream>>>(tags, sen_len, trans, feats, res);

    reduce_kernel<<<1, Bn, 0, stream>>>(res, (float*)d_out);
}
